// GATON_32865089749578
// MI455X (gfx1250) — matrix operand order repaired and checked by reading, not compiled, not run on hardware
//
#include <hip/hip_runtime.h>
#include <hip/hip_bf16.h>

#define NUM_ITEM  10000
#define NUM_SEQ   10000
#define D_IN      128
#define D_MODEL   128
#define HEADS     4
#define LAYERS    2
#define N_EDGES   400000
#define N_NODES   (NUM_ITEM + NUM_SEQ)
#define EPSV      1e-5f
#define NEG_SLOPE 0.2f
#define DHID      (HEADS * D_MODEL)   // 512

typedef __attribute__((ext_vector_type(16))) __bf16 v16bf;
typedef __attribute__((ext_vector_type(2)))  __bf16 bf16x2;
typedef __attribute__((ext_vector_type(8)))  float  v8f;
typedef __attribute__((ext_vector_type(4)))  float  fx4;
typedef __attribute__((__vector_size__(4 * sizeof(int)))) int v4i_t;

#define GLOBAL_AS __attribute__((address_space(1)))
#define LDS_AS    __attribute__((address_space(3)))

#if __has_builtin(__builtin_amdgcn_global_load_async_to_lds_b128)
#define USE_ASYNC_LDS 1
#else
#define USE_ASYNC_LDS 0
#endif

union Frag {
  v16bf    v;
  unsigned u[8];
};

// Pack two f32 -> two bf16 (RNE) into one dword (lo = a, hi = b).
__device__ inline unsigned pack2_bf16(float a, float b) {
#if __has_builtin(__builtin_amdgcn_cvt_pk_bf16_f32)
  union { bf16x2 v; unsigned u; } c;
  c.v = __builtin_amdgcn_cvt_pk_bf16_f32(a, b);   // v_cvt_pk_bf16_f32
  return c.u;
#else
  // Native bf16 ALU on gfx1250: let the backend pick hardware cvt ops.
  union { bf16x2 v; unsigned u; } c;
  c.v.x = (__bf16)a;
  c.v.y = (__bf16)b;
  return c.u;
#endif
}

// Copy 16 bytes global -> LDS. Async-DMA (ASYNCcnt) when available.
__device__ inline void copy16_g2l(const float* __restrict__ g, float* l) {
#if USE_ASYNC_LDS
  __builtin_amdgcn_global_load_async_to_lds_b128(
      (GLOBAL_AS v4i_t*)(g), (LDS_AS v4i_t*)(l), 0, 0);
#else
  *(fx4*)l = *(const fx4*)g;
#endif
}

__device__ inline void wait_async_le4() {
#if USE_ASYNC_LDS
#if __has_builtin(__builtin_amdgcn_s_wait_asynccnt)
  __builtin_amdgcn_s_wait_asynccnt(4);
#else
  asm volatile("s_wait_asynccnt 4" ::: "memory");
#endif
#endif
}
__device__ inline void wait_async_le0() {
#if USE_ASYNC_LDS
#if __has_builtin(__builtin_amdgcn_s_wait_asynccnt)
  __builtin_amdgcn_s_wait_asynccnt(0);
#else
  asm volatile("s_wait_asynccnt 0" ::: "memory");
#endif
#endif
}

// ---------------------------------------------------------------------------
// WMMA bf16 GEMM:  C[M,Nn] = A[M,K] * W[Nn,K]^T   (fp32 in/out, bf16 compute)
// Block: 256 threads = 8 waves. Tile: 64(M) x 64(N), K-step 32, double-
// buffered LDS with async-to-LDS DMA pipelined one K-step ahead.
// Wave w: rows (w&3)*16..+15, cols (w>>2)*32..+31 (two 16x16 accumulators).
// ---------------------------------------------------------------------------
#define TM 64
#define TN 64
#define TK 32
#define SA 36   // LDS row stride (floats); 144B: 16B-aligned
#define SB 36

__global__ void __launch_bounds__(256)
wmma_gemm_nt(const float* __restrict__ A, int lda,
             const float* __restrict__ W, int ldw,
             float* __restrict__ C, int ldc,
             int M, int Nn, int K)
{
  __shared__ float sA[2][TM * SA];
  __shared__ float sB[2][TN * SB];

  const int tid    = threadIdx.x;
  const int wave   = tid >> 5;
  const int lane   = tid & 31;
  const int m_band = (wave & 3) * 16;
  const int n_half = (wave >> 2) * 32;
  const int m0     = blockIdx.x * TM;
  const int n0     = blockIdx.y * TN;

  v8f acc0 = {};
  v8f acc1 = {};

  // Fragment addressing per CDNA5 ISA 7.12.2 (wave32):
  // A (16x32 bf16): lanes 0-15 -> M=lane, K base 0; lanes 16-31 -> K base 8.
  //   VGPR p holds K = (p>>2)*16 + kbase + (p&3)*2 (+1).
  const int mA     = m_band + (lane & 15);
  const int kbaseA = (lane >> 4) * 8;
  // B (32x16 bf16): same K order as A; lanes 0-15 K base 0, lanes 16-31 K base 8;
  //   column N = lane&15; VGPR p holds K = (p>>2)*16 + koff + (p&3)*2 (+1).
  const int nB0    = n_half + (lane & 15);
  const int koffB  = (lane >> 4) * 8;

  const bool mFull  = (m0 + TM <= M);
  const bool nFull  = (n0 + TN <= Nn);
  const int  nsteps = (K + TK - 1) / TK;

  // Stage one K-tile into LDS buffer `buf`. Fast path = per-lane async DMA.
  auto stage = [&](int buf, int k0) {
    const bool fast = mFull && nFull && (k0 + TK <= K);
    if (fast) {
      for (int i = tid; i < (TM * TK) / 4; i += 256) {   // 2 iters/thread
        int r = i >> 3, c4 = (i & 7) * 4;
        copy16_g2l(A + (long)(m0 + r) * lda + k0 + c4, &sA[buf][r * SA + c4]);
      }
      for (int i = tid; i < (TN * TK) / 4; i += 256) {   // 2 iters/thread
        int r = i >> 3, c4 = (i & 7) * 4;
        copy16_g2l(W + (long)(n0 + r) * ldw + k0 + c4, &sB[buf][r * SB + c4]);
      }
    } else {
      for (int i = tid; i < TM * TK; i += 256) {
        int r = i >> 5, c = i & 31;
        int gm = m0 + r, gk = k0 + c;
        bool ok = (gm < M) & (gk < K);
        float v = A[(long)(gm < M ? gm : 0) * lda + (gk < K ? gk : 0)];
        sA[buf][r * SA + c] = ok ? v : 0.0f;
      }
      for (int i = tid; i < TN * TK; i += 256) {
        int r = i >> 5, c = i & 31;
        int gn = n0 + r, gk = k0 + c;
        bool ok = (gn < Nn) & (gk < K);
        float v = W[(long)(gn < Nn ? gn : 0) * ldw + (gk < K ? gk : 0)];
        sB[buf][r * SB + c] = ok ? v : 0.0f;
      }
    }
    return fast;
  };

  // Build an 8-dword bf16 fragment from 4 contiguous-run b128 LDS loads.
  auto frag_from_runs = [](const float* r0, const float* r1,
                           const float* r2, const float* r3) {
    Frag f;
    fx4 q0 = *(const fx4*)r0;
    fx4 q1 = *(const fx4*)r1;
    fx4 q2 = *(const fx4*)r2;
    fx4 q3 = *(const fx4*)r3;
    f.u[0] = pack2_bf16(q0.x, q0.y); f.u[1] = pack2_bf16(q0.z, q0.w);
    f.u[2] = pack2_bf16(q1.x, q1.y); f.u[3] = pack2_bf16(q1.z, q1.w);
    f.u[4] = pack2_bf16(q2.x, q2.y); f.u[5] = pack2_bf16(q2.z, q2.w);
    f.u[6] = pack2_bf16(q3.x, q3.y); f.u[7] = pack2_bf16(q3.z, q3.w);
    return f;
  };

  stage(0, 0);

  for (int s = 0; s < nsteps; ++s) {
    const int cur = s & 1;
    bool nextFast = false;
    if (s + 1 < nsteps) nextFast = stage(1 - cur, (s + 1) * TK);

    // ASYNCcnt completes in-order per wave: if the next tile's 4 chunk-loads
    // are the only outstanding ones, the current tile has landed.
    if (nextFast) wait_async_le4(); else wait_async_le0();
    __syncthreads();

    // A fragment: pairs at kbase+{0..7} and kbase+16+{0..7} -> 4x ds_load_b128
    const float* abase = &sA[cur][mA * SA + kbaseA];
    Frag a = frag_from_runs(abase, abase + 4, abase + 16, abase + 20);
    // B fragments: pairs at koff+{0..7} and koff+16+{0..7} -> 4x ds_load_b128 each
    const float* bbase0 = &sB[cur][nB0 * SB + koffB];
    Frag b0 = frag_from_runs(bbase0, bbase0 + 4, bbase0 + 16, bbase0 + 20);
    const float* bbase1 = &sB[cur][(nB0 + 16) * SB + koffB];
    Frag b1 = frag_from_runs(bbase1, bbase1 + 4, bbase1 + 16, bbase1 + 20);

    acc0 = __builtin_amdgcn_wmma_f32_16x16x32_bf16(
        false, a.v, false, b0.v, (short)0, acc0, false, false);
    acc1 = __builtin_amdgcn_wmma_f32_16x16x32_bf16(
        false, a.v, false, b1.v, (short)0, acc1, false, false);

    __syncthreads();   // all waves done reading buf[cur] before it is refilled
  }

  // C 16x16 f32 layout: VGPR p -> row p (lanes 0-15) / row p+8 (lanes 16-31),
  // col = lane&15.
  const int ncol0  = n0 + nB0;
  const int rowsel = (lane >> 4) * 8;
#pragma unroll
  for (int p = 0; p < 8; ++p) {
    int m = m0 + m_band + p + rowsel;
    if (m < M) {
      if (ncol0 < Nn)      C[(long)m * ldc + ncol0]      = acc0[p];
      if (ncol0 + 16 < Nn) C[(long)m * ldc + ncol0 + 16] = acc1[p];
    }
  }
}

// ---------------------------------------------------------------------------
// alpha_s / alpha_d : per-node, per-head dot of X[n,h,:] with att vectors.
// One block (128 threads) per node.
// ---------------------------------------------------------------------------
__global__ void __launch_bounds__(128)
alpha_kernel(const float* __restrict__ X,
             const float* __restrict__ a_src,
             const float* __restrict__ a_dst,
             float* __restrict__ AS, float* __restrict__ AD)
{
  __shared__ float red[128];
  const int n = blockIdx.x;
  const int t = threadIdx.x;
#pragma unroll
  for (int h = 0; h < HEADS; ++h) {
    const float x = X[(long)n * DHID + h * D_MODEL + t];
    red[t] = x * a_src[h * D_MODEL + t];
    __syncthreads();
    for (int s = 64; s > 0; s >>= 1) {
      if (t < s) red[t] += red[t + s];
      __syncthreads();
    }
    if (t == 0) AS[n * HEADS + h] = red[0];
    __syncthreads();
    red[t] = x * a_dst[h * D_MODEL + t];
    __syncthreads();
    for (int s = 64; s > 0; s >>= 1) {
      if (t < s) red[t] += red[t + s];
      __syncthreads();
    }
    if (t == 0) AD[n * HEADS + h] = red[0];
    __syncthreads();
  }
}

__global__ void fill_kernel(float* __restrict__ p, float v, int count) {
  int i = blockIdx.x * blockDim.x + threadIdx.x;
  if (i < count) p[i] = v;
}

__device__ inline void atomicMaxF(float* addr, float val) {
  unsigned* u = (unsigned*)addr;
  unsigned old = *u;
  while (__uint_as_float(old) < val) {
    unsigned assumed = old;
    old = atomicCAS(u, assumed, __float_as_uint(val));
    if (old == assumed) break;
  }
}

// Pass 1: leaky-relu logits + segment max (dst).
__global__ void edge_max_kernel(const int* __restrict__ src, const int* __restrict__ dst,
                                const float* __restrict__ AS, const float* __restrict__ AD,
                                float* __restrict__ EV, float* __restrict__ MX,
                                int E, int Ntot)
{
  int e = blockIdx.x * blockDim.x + threadIdx.x;
  if (e >= E + Ntot) return;
  int s, d;
  if (e < E) { s = src[e]; d = dst[e]; } else { s = d = e - E; }  // self loops
#pragma unroll
  for (int h = 0; h < HEADS; ++h) {
    float v = AS[s * HEADS + h] + AD[d * HEADS + h];
    v = (v > 0.0f) ? v : NEG_SLOPE * v;
    EV[(long)e * HEADS + h] = v;
    atomicMaxF(&MX[d * HEADS + h], v);
  }
}

// Pass 2: exp(e - max) + segment sum.
__global__ void edge_exp_kernel(const int* __restrict__ dst,
                                const float* __restrict__ MX,
                                float* __restrict__ EV, float* __restrict__ Z,
                                int E, int Ntot)
{
  int e = blockIdx.x * blockDim.x + threadIdx.x;
  if (e >= E + Ntot) return;
  int d = (e < E) ? dst[e] : (e - E);
#pragma unroll
  for (int h = 0; h < HEADS; ++h) {
    float v = __expf(EV[(long)e * HEADS + h] - MX[d * HEADS + h]);
    EV[(long)e * HEADS + h] = v;
    atomicAdd(&Z[d * HEADS + h], v);
  }
}

// Pass 3: weighted message scatter. One block (128 threads) per edge.
__global__ void __launch_bounds__(128)
edge_scatter_kernel(const int* __restrict__ src, const int* __restrict__ dst,
                    const float* __restrict__ X, const float* __restrict__ EV,
                    const float* __restrict__ Z, float* __restrict__ AGG,
                    int E, int Ntot)
{
  int e = blockIdx.x;
  if (e >= E + Ntot) return;
  int s, d;
  if (e < E) { s = src[e]; d = dst[e]; } else { s = d = e - E; }
  const int t = threadIdx.x;
#pragma unroll
  for (int h = 0; h < HEADS; ++h) {
    float a = EV[(long)e * HEADS + h] / Z[d * HEADS + h];
    atomicAdd(&AGG[(long)d * DHID + h * D_MODEL + t],
              X[(long)s * DHID + h * D_MODEL + t] * a);
  }
}

// Mean over heads + bias + ReLU.
__global__ void mean_bias_relu_kernel(const float* __restrict__ AGG,
                                      const float* __restrict__ bias,
                                      float* __restrict__ Hout, int Ntot)
{
  int i = blockIdx.x * blockDim.x + threadIdx.x;
  if (i >= Ntot * D_MODEL) return;
  int n = i >> 7, o = i & 127;
  float v = 0.0f;
#pragma unroll
  for (int h = 0; h < HEADS; ++h) v += AGG[(long)n * DHID + h * D_MODEL + o];
  v = v * (1.0f / HEADS) + bias[o];
  Hout[i] = fmaxf(v, 0.0f);
}

// BatchNorm (affine=False), batch stats. One block per column.
__global__ void __launch_bounds__(256)
bn_stats_kernel(const float* __restrict__ Hin, float* __restrict__ stat, int Ntot)
{
  __shared__ float ssum[256];
  __shared__ float ssq[256];
  const int o = blockIdx.x;
  const int t = threadIdx.x;
  float s = 0.0f, q = 0.0f;
  for (int n = t; n < Ntot; n += 256) {
    float v = Hin[(long)n * D_MODEL + o];
    s += v; q += v * v;
  }
  ssum[t] = s; ssq[t] = q;
  __syncthreads();
  for (int w = 128; w > 0; w >>= 1) {
    if (t < w) { ssum[t] += ssum[t + w]; ssq[t] += ssq[t + w]; }
    __syncthreads();
  }
  if (t == 0) {
    float mu = ssum[0] / (float)Ntot;
    stat[o] = mu;
    stat[D_MODEL + o] = ssq[0] / (float)Ntot - mu * mu;  // biased var
  }
}

__global__ void bn_norm_kernel(const float* __restrict__ Hin,
                               const float* __restrict__ stat,
                               float* __restrict__ Hout, int Ntot)
{
  int i = blockIdx.x * blockDim.x + threadIdx.x;
  if (i >= Ntot * D_MODEL) return;
  int o = i & 127;
  Hout[i] = (Hin[i] - stat[o]) * rsqrtf(stat[D_MODEL + o] + EPSV);
}

// ---------------------------------------------------------------------------
extern "C" void kernel_launch(void* const* d_in, const int* in_sizes, int n_in,
                              void* d_out, int out_size, void* d_ws, size_t ws_size,
                              hipStream_t stream)
{
  const float* x_item  = (const float*)d_in[0];
  const float* x_seq   = (const float*)d_in[1];
  const int*   eidx    = (const int*)  d_in[2];
  const float* W_item  = (const float*)d_in[3];
  const float* W_seq   = (const float*)d_in[4];
  const float* lin_w   = (const float*)d_in[5];
  const float* att_src = (const float*)d_in[6];
  const float* att_dst = (const float*)d_in[7];
  const float* bias    = (const float*)d_in[8];
  float* out = (float*)d_out;

  const int* src  = eidx;
  const int* dst  = eidx + N_EDGES;
  const int  Etot = N_EDGES + N_NODES;

  // Workspace carve-up (floats). ~100 MB total.
  float* ws   = (float*)d_ws;
  float* H    = ws;                                   // N_NODES * 128
  float* X    = H    + (size_t)N_NODES * D_MODEL;     // N_NODES * 512
  float* AGG  = X    + (size_t)N_NODES * DHID;        // N_NODES * 512
  float* AS   = AGG  + (size_t)N_NODES * DHID;        // N_NODES * 4
  float* AD   = AS   + (size_t)N_NODES * HEADS;
  float* MX   = AD   + (size_t)N_NODES * HEADS;
  float* Z    = MX   + (size_t)N_NODES * HEADS;
  float* EV   = Z    + (size_t)N_NODES * HEADS;       // Etot * 4
  float* STAT = EV   + (size_t)Etot * HEADS;          // 256

  const dim3 blk(256);

  // Input projections -> H (concat of h_item, h_seq).
  {
    dim3 g((NUM_ITEM + TM - 1) / TM, (D_MODEL + TN - 1) / TN);
    wmma_gemm_nt<<<g, blk, 0, stream>>>(x_item, D_IN, W_item, D_IN,
                                        H, D_MODEL, NUM_ITEM, D_MODEL, D_IN);
  }
  {
    dim3 g((NUM_SEQ + TM - 1) / TM, (D_MODEL + TN - 1) / TN);
    wmma_gemm_nt<<<g, blk, 0, stream>>>(x_seq, NUM_ITEM, W_seq, NUM_ITEM,
                                        H + (size_t)NUM_ITEM * D_MODEL, D_MODEL,
                                        NUM_SEQ, D_MODEL, NUM_ITEM);
  }

  const int nh     = N_NODES * HEADS;
  const int aggN   = N_NODES * DHID;
  const int hElems = N_NODES * D_MODEL;

  for (int l = 0; l < LAYERS; ++l) {
    // X = H @ lin_w[l]^T   [N, 512]
    {
      dim3 g((N_NODES + TM - 1) / TM, (DHID + TN - 1) / TN);
      wmma_gemm_nt<<<g, blk, 0, stream>>>(H, D_MODEL,
                                          lin_w + (size_t)l * DHID * D_MODEL, D_MODEL,
                                          X, DHID, N_NODES, DHID, D_MODEL);
    }
    alpha_kernel<<<N_NODES, 128, 0, stream>>>(X,
        att_src + (size_t)l * HEADS * D_MODEL,
        att_dst + (size_t)l * HEADS * D_MODEL, AS, AD);

    fill_kernel<<<(nh + 255) / 256, 256, 0, stream>>>(MX, -__builtin_inff(), nh);
    fill_kernel<<<(nh + 255) / 256, 256, 0, stream>>>(Z, 0.0f, nh);
    fill_kernel<<<(aggN + 255) / 256, 256, 0, stream>>>(AGG, 0.0f, aggN);

    edge_max_kernel<<<(Etot + 255) / 256, 256, 0, stream>>>(src, dst, AS, AD, EV, MX,
                                                            N_EDGES, N_NODES);
    edge_exp_kernel<<<(Etot + 255) / 256, 256, 0, stream>>>(dst, MX, EV, Z,
                                                            N_EDGES, N_NODES);
    edge_scatter_kernel<<<Etot, 128, 0, stream>>>(src, dst, X, EV, Z, AGG,
                                                  N_EDGES, N_NODES);

    mean_bias_relu_kernel<<<(hElems + 255) / 256, 256, 0, stream>>>(
        AGG, bias + (size_t)l * D_MODEL, H, N_NODES);

    bn_stats_kernel<<<D_MODEL, 256, 0, stream>>>(H, STAT, N_NODES);
    float* dstH = (l == LAYERS - 1) ? out : H;
    bn_norm_kernel<<<(hElems + 255) / 256, 256, 0, stream>>>(H, STAT, dstH, N_NODES);
  }
}
